// TemporalBlock_42872363548704
// MI455X (gfx1250) — compile-verified, matrix operand order checked by reading, not run on hardware
//
#include <hip/hip_runtime.h>
#include <math.h>
#include <stdint.h>

// ---------------------------------------------------------------------------
// Types for CDNA5 WMMA (wave32): v_wmma_f32_16x16x32_bf16
// ---------------------------------------------------------------------------
typedef __attribute__((ext_vector_type(16))) __bf16 v16bf;
typedef __attribute__((ext_vector_type(8)))  __bf16 v8bf;
typedef __attribute__((ext_vector_type(8)))  float  v8f;

__device__ __forceinline__ v8f wmma_bf16(v16bf a, v16bf b, v8f c) {
  return __builtin_amdgcn_wmma_f32_16x16x32_bf16(false, a, false, b, (short)0, c,
                                                 false, false);
}

// A-fragment per CDNA5 ISA 16-bit A layout (16x32):
//  lanes 0-15 : row = lane,    K = {0..7, 16..23} of the 32-chunk
//  lanes 16-31: row = lane-16, K = {8..15, 24..31}
// p points at &row[k0] where k0 = (lane>=16) ? 8 : 0; chunks are 16B aligned.
__device__ __forceinline__ v16bf ldA(const __bf16* p) {
  v8bf lo = *(const v8bf*)(p);
  v8bf hi = *(const v8bf*)(p + 16);
  return __builtin_shufflevector(lo, hi, 0, 1, 2, 3, 4, 5, 6, 7,
                                 8, 9, 10, 11, 12, 13, 14, 15);
}

__device__ __forceinline__ float eluf(float v) {
  return v > 0.f ? v : (__expf(v) - 1.f);
}
__device__ __forceinline__ float sigf(float v) {
  return 1.f / (1.f + __expf(-v));
}

#define HID 256
#define JDIM 32
#define BATCH 32
#define TLEN 128
#define MROWS 4096  // B*T

// ---------------------------------------------------------------------------
// Pre-swizzle a row-major fp32 [K,N] weight into bf16 WMMA B-fragment layout.
// Fragment element e = ((kt*(N/16)+nt)*32 + lane)*16 + i, mirroring the A
// layout with M->N so each lane consumes 32 contiguous bytes per fragment.
// ---------------------------------------------------------------------------
__global__ void swizzle_bf16(const float* __restrict__ src,
                             __bf16* __restrict__ dst, int K, int N) {
  int e = blockIdx.x * 256 + threadIdx.x;
  if (e >= K * N) return;
  int i = e & 15;
  int lane = (e >> 4) & 31;
  int rest = e >> 9;
  int ntiles = N >> 4;
  int nt = rest % ntiles;
  int kt = rest / ntiles;
  int hi = lane >> 4, ln = lane & 15;
  int n = nt * 16 + ln;
  int kk = (hi ? 8 : 0) + ((i < 8) ? i : 16 + (i - 8));
  int k = kt * 32 + kk;
  dst[e] = (__bf16)src[(size_t)k * N + n];
}

// ---------------------------------------------------------------------------
// Per-asset embedding matvecs: ev_ws = vsn_emb[sid]@ws_w+ws_b,
// h0/c0 = elu(emb[sid]@w1+b1)@w2+b2, ef = ffn_emb[sid]@l2_w+l2_b.
// ---------------------------------------------------------------------------
__global__ __launch_bounds__(256) void prep_assets(
    const int* __restrict__ sid, const float* __restrict__ vsn_emb,
    const float* __restrict__ ws_w, const float* __restrict__ ws_b,
    const float* __restrict__ emb, const float* __restrict__ h0_w1,
    const float* __restrict__ h0_b1, const float* __restrict__ h0_w2,
    const float* __restrict__ h0_b2, const float* __restrict__ c0_w1,
    const float* __restrict__ c0_b1, const float* __restrict__ c0_w2,
    const float* __restrict__ c0_b2, const float* __restrict__ ffn_emb,
    const float* __restrict__ l2_w, const float* __restrict__ l2_b,
    float* __restrict__ ev_ws, float* __restrict__ h0o,
    float* __restrict__ c0o, float* __restrict__ efo) {
  __shared__ float e_row[HID];
  __shared__ float tmp[HID];
  int b = blockIdx.x, tid = threadIdx.x;
  int sb = sid[b];

  e_row[tid] = vsn_emb[(size_t)sb * HID + tid];
  __syncthreads();
  {
    float s = ws_b[tid];
    for (int k = 0; k < HID; ++k) s += e_row[k] * ws_w[k * HID + tid];
    ev_ws[b * HID + tid] = s;
  }
  __syncthreads();
  e_row[tid] = emb[(size_t)sb * HID + tid];
  __syncthreads();
  {
    float s = h0_b1[tid];
    for (int k = 0; k < HID; ++k) s += e_row[k] * h0_w1[k * HID + tid];
    tmp[tid] = eluf(s);
  }
  __syncthreads();
  {
    float s = h0_b2[tid];
    for (int k = 0; k < HID; ++k) s += tmp[k] * h0_w2[k * HID + tid];
    h0o[b * HID + tid] = s;
  }
  __syncthreads();
  {
    float s = c0_b1[tid];
    for (int k = 0; k < HID; ++k) s += e_row[k] * c0_w1[k * HID + tid];
    tmp[tid] = eluf(s);
  }
  __syncthreads();
  {
    float s = c0_b2[tid];
    for (int k = 0; k < HID; ++k) s += tmp[k] * c0_w2[k * HID + tid];
    c0o[b * HID + tid] = s;
  }
  __syncthreads();
  e_row[tid] = ffn_emb[(size_t)sb * HID + tid];
  __syncthreads();
  {
    float s = l2_b[tid];
    for (int k = 0; k < HID; ++k) s += e_row[k] * l2_w[k * HID + tid];
    efo[b * HID + tid] = s;
  }
}

// ---------------------------------------------------------------------------
// VSN gating: w = softmax(elu(x@wx_w + wx_b + ev_ws) @ wout_w + wout_b)
// ---------------------------------------------------------------------------
__global__ __launch_bounds__(256) void gate_softmax(
    const float* __restrict__ x, const float* __restrict__ ev_ws,
    const float* __restrict__ wx_w, const float* __restrict__ wx_b,
    const float* __restrict__ wout_w, const float* __restrict__ wout_b,
    float* __restrict__ w_sm) {
  __shared__ float xr[JDIM];
  __shared__ float gh[HID];
  __shared__ float lg[JDIM];
  __shared__ float ex[JDIM];
  int row = blockIdx.x, tid = threadIdx.x;
  if (tid < JDIM) xr[tid] = x[(size_t)row * JDIM + tid];
  __syncthreads();
  int b = row >> 7;
  float s = wx_b[tid] + ev_ws[b * HID + tid];
  for (int j = 0; j < JDIM; ++j) s += xr[j] * wx_w[j * HID + tid];
  gh[tid] = eluf(s);
  __syncthreads();
  if (tid < JDIM) {
    float l = wout_b[tid];
    for (int h = 0; h < HID; ++h) l += gh[h] * wout_w[h * JDIM + tid];
    lg[tid] = l;
  }
  __syncthreads();
  if (tid < JDIM) {
    float mx = lg[0];
    for (int j = 1; j < JDIM; ++j) mx = fmaxf(mx, lg[j]);
    ex[tid] = __expf(lg[tid] - mx);
  }
  __syncthreads();
  if (tid < JDIM) {
    float sm = 0.f;
    for (int j = 0; j < JDIM; ++j) sm += ex[j];
    w_sm[(size_t)row * JDIM + tid] = ex[tid] / sm;
  }
}

// ---------------------------------------------------------------------------
// Dominant kernel: x_p = LN( u @ f2w_flat + sum_j w_j*f2b_j )
// u[row, j*256+h] = w[row,j] * elu(x[row,j]*f1w[j,h]+f1b[j,h]) generated
// on the fly into a DOUBLE-BUFFERED bf16 LDS tile (1 barrier per K-chunk);
// bias folded as K-chunk 256. x/w tiles staged via async global->LDS copy.
// Block = 32 rows x 256 cols, 8 waves (2 Mgroups x 4 Ngroups), K = 257*32.
// ---------------------------------------------------------------------------
__global__ __launch_bounds__(256) void vsn_gemm(
    const float* __restrict__ x, const float* __restrict__ w_sm,
    const float* __restrict__ f1w, const float* __restrict__ f1b,
    const __bf16* __restrict__ Bfrag, const float* __restrict__ ln_g,
    const float* __restrict__ ln_b, float* __restrict__ x_p) {
  __shared__ __align__(16) float x_t[32][JDIM];
  __shared__ __align__(16) float w_t[32][JDIM];
  __shared__ __align__(32) __bf16 A_lds[2][32][32];
  __shared__ float xp[32][260];
  __shared__ float ps[32][8], pq[32][8];
  __shared__ float stats[32][2];

  int tid = threadIdx.x;
  int m0 = blockIdx.x * 32;
  int lane = tid & 31, wv = tid >> 5;
  int mg = wv >> 2, ng = wv & 3;
  int hi = lane >> 4, ln = lane & 15;

  // --- async global->LDS staging of the contiguous 4KB x / w tiles ---
  // (GLOBAL_LOAD_ASYNC_TO_LDS_B128, tracked by ASYNCcnt)
  {
    unsigned ldsx = (unsigned)(uintptr_t)(&x_t[0][0]) + (unsigned)tid * 16u;
    unsigned ldsw = (unsigned)(uintptr_t)(&w_t[0][0]) + (unsigned)tid * 16u;
    unsigned long long gx =
        (unsigned long long)(uintptr_t)(x + (size_t)m0 * JDIM) +
        (unsigned long long)tid * 16u;
    unsigned long long gw =
        (unsigned long long)(uintptr_t)(w_sm + (size_t)m0 * JDIM) +
        (unsigned long long)tid * 16u;
    asm volatile("global_load_async_to_lds_b128 %0, %1, off" ::"v"(ldsx),
                 "v"(gx)
                 : "memory");
    asm volatile("global_load_async_to_lds_b128 %0, %1, off" ::"v"(ldsw),
                 "v"(gw)
                 : "memory");
    asm volatile("s_wait_asynccnt 0x0" ::: "memory");
  }
  __syncthreads();

  int genr = tid >> 3, genc0 = (tid & 7) * 4;
  auto gen = [&](int kt, int buf) {
    if (kt < 256) {
      int j = kt >> 3, kkb = (kt & 7) * 32;
      float wj = w_t[genr][j], xv = x_t[genr][j];
      const float* fw = f1w + j * HID + kkb + genc0;
      const float* fb = f1b + j * HID + kkb + genc0;
      for (int c = 0; c < 4; ++c)
        A_lds[buf][genr][genc0 + c] = (__bf16)(wj * eluf(xv * fw[c] + fb[c]));
    } else {  // bias chunk: A = softmax weights, B = f2_b
      for (int c = 0; c < 4; ++c)
        A_lds[buf][genr][genc0 + c] = (__bf16)w_t[genr][genc0 + c];
    }
  };

  v8f acc[4] = {};
  gen(0, 0);
  __syncthreads();

  for (int kt = 0; kt < 257; ++kt) {
    int cur = kt & 1;
    if (kt + 1 < 257) gen(kt + 1, cur ^ 1);  // overlap VALU gen with WMMA pipe
    if (kt + 2 < 257) {                      // prefetch B fragment stream
      const char* pf = (const char*)((const v16bf*)Bfrag +
                                     (((size_t)(kt + 2) * 16 + ng * 4) * 32 +
                                      lane));
      __builtin_prefetch(pf, 0, 1);
    }
    v16bf af = ldA(&A_lds[cur][mg * 16 + ln][hi ? 8 : 0]);
    const v16bf* bp = (const v16bf*)Bfrag + ((size_t)kt * 16) * 32 + lane;
    for (int t = 0; t < 4; ++t) {
      int nt = ng * 4 + t;
      v16bf bf = bp[(size_t)nt * 32];
      acc[t] = wmma_bf16(af, bf, acc[t]);
    }
    __syncthreads();
  }

  // ---- epilogue: LayerNorm over H=256 per row ----
  for (int t = 0; t < 4; ++t)
    for (int r = 0; r < 8; ++r) {
      int ml = mg * 16 + r + 8 * hi;
      int n = ng * 64 + t * 16 + ln;
      xp[ml][n] = acc[t][r];
    }
  __syncthreads();
  int row = tid >> 3, l8 = tid & 7;
  float s = 0.f, q = 0.f;
  for (int c = l8 * 32; c < l8 * 32 + 32; ++c) {
    float v = xp[row][c];
    s += v;
    q += v * v;
  }
  ps[row][l8] = s;
  pq[row][l8] = q;
  __syncthreads();
  if (l8 == 0) {
    float S = 0.f, Q = 0.f;
    for (int i = 0; i < 8; ++i) {
      S += ps[row][i];
      Q += pq[row][i];
    }
    float mean = S * (1.f / 256.f);
    float var = Q * (1.f / 256.f) - mean * mean;
    stats[row][0] = mean;
    stats[row][1] = rsqrtf(var + 1e-5f);
  }
  __syncthreads();
  float mean = stats[row][0], rinv = stats[row][1];
  for (int c = l8 * 32; c < l8 * 32 + 32; ++c)
    x_p[(size_t)(m0 + row) * HID + c] =
        (xp[row][c] - mean) * rinv * ln_g[c] + ln_b[c];
}

// ---------------------------------------------------------------------------
// Generic K=256 WMMA GEMM: out = epilogue(A[M,256] @ Bfrag + bias).
// EPI 0: + bias, store fp32 (out ld = N)                   -> xg
// EPI 1: + bias + rowadd[b], elu, store fp32 (N=256)       -> ffn hidden
// EPI 2: + bias + resid, LayerNorm(g,b), store fp32 (N=256)-> final out
// ---------------------------------------------------------------------------
template <int EPI>
__global__ __launch_bounds__(256) void gemm_k256(
    const float* __restrict__ A, const __bf16* __restrict__ Bfrag, int N,
    const float* __restrict__ bias, const float* __restrict__ rowadd,
    const float* __restrict__ resid, const float* __restrict__ ln_g,
    const float* __restrict__ ln_b, float* __restrict__ out) {
  __shared__ __align__(32) __bf16 A_lds[32][HID];
  __shared__ float xp[32][260];
  __shared__ float ps[32][8], pq[32][8];
  __shared__ float stats[32][2];

  int tid = threadIdx.x;
  int m0 = blockIdx.x * 32;
  int col0 = blockIdx.y * 256;
  int lane = tid & 31, wv = tid >> 5;
  int mg = wv >> 2, ng = wv & 3;
  int hi = lane >> 4, ln = lane & 15;
  int ntiles = N >> 4;

  for (int idx = tid; idx < 32 * HID; idx += 256) {
    int r = idx >> 8, c = idx & 255;
    A_lds[r][c] = (__bf16)A[(size_t)(m0 + r) * HID + c];
  }
  __syncthreads();

  v8f acc[4] = {};
  for (int kt = 0; kt < 8; ++kt) {
    if (kt + 1 < 8) {
      const char* pf = (const char*)((const v16bf*)Bfrag +
                                     (((size_t)(kt + 1) * ntiles + (col0 >> 4) +
                                       ng * 4) *
                                          32 +
                                      lane));
      __builtin_prefetch(pf, 0, 1);
    }
    v16bf af = ldA(&A_lds[mg * 16 + ln][kt * 32 + (hi ? 8 : 0)]);
    for (int t = 0; t < 4; ++t) {
      int nt = (col0 >> 4) + ng * 4 + t;
      const v16bf* bp =
          (const v16bf*)Bfrag + ((size_t)(kt * ntiles + nt) * 32 + lane);
      acc[t] = wmma_bf16(af, *bp, acc[t]);
    }
  }

  if (EPI == 0) {
    for (int t = 0; t < 4; ++t)
      for (int r = 0; r < 8; ++r) {
        int m = m0 + mg * 16 + r + 8 * hi;
        int n = col0 + ng * 64 + t * 16 + ln;
        out[(size_t)m * N + n] = acc[t][r] + bias[n];
      }
  } else if (EPI == 1) {
    for (int t = 0; t < 4; ++t)
      for (int r = 0; r < 8; ++r) {
        int m = m0 + mg * 16 + r + 8 * hi;
        int n = ng * 64 + t * 16 + ln;
        float v = acc[t][r] + bias[n] + rowadd[(m >> 7) * HID + n];
        out[(size_t)m * HID + n] = eluf(v);
      }
  } else {
    for (int t = 0; t < 4; ++t)
      for (int r = 0; r < 8; ++r) {
        int ml = mg * 16 + r + 8 * hi;
        int n = ng * 64 + t * 16 + ln;
        xp[ml][n] = acc[t][r] + bias[n] + resid[(size_t)(m0 + ml) * HID + n];
      }
    __syncthreads();
    int row = tid >> 3, l8 = tid & 7;
    float s = 0.f, q = 0.f;
    for (int c = l8 * 32; c < l8 * 32 + 32; ++c) {
      float v = xp[row][c];
      s += v;
      q += v * v;
    }
    ps[row][l8] = s;
    pq[row][l8] = q;
    __syncthreads();
    if (l8 == 0) {
      float S = 0.f, Q = 0.f;
      for (int i = 0; i < 8; ++i) {
        S += ps[row][i];
        Q += pq[row][i];
      }
      float mean = S * (1.f / 256.f);
      float var = Q * (1.f / 256.f) - mean * mean;
      stats[row][0] = mean;
      stats[row][1] = rsqrtf(var + 1e-5f);
    }
    __syncthreads();
    float mean = stats[row][0], rinv = stats[row][1];
    for (int c = l8 * 32; c < l8 * 32 + 32; ++c)
      out[(size_t)(m0 + row) * HID + c] =
          (xp[row][c] - mean) * rinv * ln_g[c] + ln_b[c];
  }
}

// ---------------------------------------------------------------------------
// LSTM recurrence: single persistent workgroup, 16 waves.
// Wave wv owns gate-space columns [wv*16, wv*16+16) for ALL 4 gates and both
// 16-row groups: 8 accumulator tiles -> gate nonlinearity needs no cross-wave
// exchange. Cell state c lives in registers; h ping-pongs through LDS (bf16).
// ---------------------------------------------------------------------------
__global__ __launch_bounds__(512) void lstm_scan(
    const __bf16* __restrict__ Whh_frag, const float* __restrict__ xg,
    const float* __restrict__ bhh, const float* __restrict__ h0,
    const float* __restrict__ c0, float* __restrict__ h_seq) {
  __shared__ __align__(32) __bf16 h_lds[BATCH][HID];
  int tid = threadIdx.x;
  int lane = tid & 31, wv = tid >> 5;  // wv in [0,16)
  int hi = lane >> 4, ln = lane & 15;

  for (int idx = tid; idx < BATCH * HID; idx += 512)
    h_lds[idx >> 8][idx & 255] = (__bf16)h0[idx];

  float c_reg[16];
  for (int mg = 0; mg < 2; ++mg)
    for (int r = 0; r < 8; ++r) {
      int b = mg * 16 + r + 8 * hi;
      c_reg[mg * 8 + r] = c0[b * HID + wv * 16 + ln];
    }
  __syncthreads();

  for (int t = 0; t < TLEN; ++t) {
    v8f acc[2][4] = {};
    for (int kt = 0; kt < 8; ++kt) {
      v16bf af[2];
      for (int mg = 0; mg < 2; ++mg)
        af[mg] = ldA(&h_lds[mg * 16 + ln][kt * 32 + (hi ? 8 : 0)]);
      for (int g = 0; g < 4; ++g) {
        int nt = g * 16 + wv;  // N=1024 -> 64 n-tiles
        const v16bf* bp =
            (const v16bf*)Whh_frag + ((size_t)(kt * 64 + nt) * 32 + lane);
        v16bf bf = *bp;
        acc[0][g] = wmma_bf16(af[0], bf, acc[0][g]);
        acc[1][g] = wmma_bf16(af[1], bf, acc[1][g]);
      }
    }
    __syncthreads();  // all h_lds reads complete before overwrite
    for (int mg = 0; mg < 2; ++mg)
      for (int r = 0; r < 8; ++r) {
        int b = mg * 16 + r + 8 * hi;
        int n = wv * 16 + ln;
        const float* xgr = xg + ((size_t)(b * TLEN + t)) * 1024;
        float iv = acc[mg][0][r] + xgr[n] + bhh[n];
        float fv = acc[mg][1][r] + xgr[256 + n] + bhh[256 + n];
        float gv = acc[mg][2][r] + xgr[512 + n] + bhh[512 + n];
        float ov = acc[mg][3][r] + xgr[768 + n] + bhh[768 + n];
        float c = sigf(fv) * c_reg[mg * 8 + r] + sigf(iv) * tanhf(gv);
        c_reg[mg * 8 + r] = c;
        float h = sigf(ov) * tanhf(c);
        h_seq[((size_t)(b * TLEN + t)) * HID + n] = h;
        h_lds[b][n] = (__bf16)h;
      }
    __syncthreads();
  }
}

// ---------------------------------------------------------------------------
// a = LN(h_seq + x_p): one block per (b,t) row.
// ---------------------------------------------------------------------------
__global__ __launch_bounds__(256) void residual_ln(
    const float* __restrict__ A, const float* __restrict__ Bv,
    const float* __restrict__ g, const float* __restrict__ bb,
    float* __restrict__ out) {
  __shared__ float s1[256], s2[256];
  int row = blockIdx.x, tid = threadIdx.x;
  float v = A[(size_t)row * HID + tid] + Bv[(size_t)row * HID + tid];
  s1[tid] = v;
  s2[tid] = v * v;
  __syncthreads();
  for (int off = 128; off > 0; off >>= 1) {
    if (tid < off) {
      s1[tid] += s1[tid + off];
      s2[tid] += s2[tid + off];
    }
    __syncthreads();
  }
  float mean = s1[0] * (1.f / 256.f);
  float var = s2[0] * (1.f / 256.f) - mean * mean;
  out[(size_t)row * HID + tid] =
      (v - mean) * rsqrtf(var + 1e-5f) * g[tid] + bb[tid];
}

// ---------------------------------------------------------------------------
extern "C" void kernel_launch(void* const* d_in, const int* in_sizes, int n_in,
                              void* d_out, int out_size, void* d_ws,
                              size_t ws_size, hipStream_t stream) {
  const float* x        = (const float*)d_in[0];
  const int*   sid      = (const int*)d_in[1];
  const float* vsn_emb  = (const float*)d_in[2];
  const float* wx_w     = (const float*)d_in[3];
  const float* wx_b     = (const float*)d_in[4];
  const float* ws_w     = (const float*)d_in[5];
  const float* ws_b     = (const float*)d_in[6];
  const float* wout_w   = (const float*)d_in[7];
  const float* wout_b   = (const float*)d_in[8];
  const float* f1w      = (const float*)d_in[9];
  const float* f1b      = (const float*)d_in[10];
  const float* f2w      = (const float*)d_in[11];
  const float* f2b      = (const float*)d_in[12];
  const float* vnorm_g  = (const float*)d_in[13];
  const float* vnorm_b  = (const float*)d_in[14];
  const float* emb      = (const float*)d_in[15];
  const float* h0_w1    = (const float*)d_in[16];
  const float* h0_b1    = (const float*)d_in[17];
  const float* h0_w2    = (const float*)d_in[18];
  const float* h0_b2    = (const float*)d_in[19];
  const float* c0_w1    = (const float*)d_in[20];
  const float* c0_b1    = (const float*)d_in[21];
  const float* c0_w2    = (const float*)d_in[22];
  const float* c0_b2    = (const float*)d_in[23];
  const float* wih      = (const float*)d_in[24];
  const float* whh      = (const float*)d_in[25];
  const float* bih      = (const float*)d_in[26];
  const float* bhh      = (const float*)d_in[27];
  const float* n1_g     = (const float*)d_in[28];
  const float* n1_b     = (const float*)d_in[29];
  const float* ffn_emb  = (const float*)d_in[30];
  const float* l1_w     = (const float*)d_in[31];
  const float* l1_b     = (const float*)d_in[32];
  const float* l2_w     = (const float*)d_in[33];
  const float* l2_b     = (const float*)d_in[34];
  const float* l3_w     = (const float*)d_in[35];
  const float* l3_b     = (const float*)d_in[36];
  const float* n2_g     = (const float*)d_in[37];
  const float* n2_b     = (const float*)d_in[38];

  size_t off = 0;
  auto alloc = [&](size_t bytes) -> void* {
    void* p = (char*)d_ws + off;
    off = (off + bytes + 255) & ~(size_t)255;
    return p;
  };
  float*  ev_ws  = (float*)alloc(32 * 256 * 4);
  float*  h0w    = (float*)alloc(32 * 256 * 4);
  float*  c0w    = (float*)alloc(32 * 256 * 4);
  float*  efw    = (float*)alloc(32 * 256 * 4);
  float*  w_sm   = (float*)alloc((size_t)4096 * 32 * 4);
  float*  x_p    = (float*)alloc((size_t)4096 * 256 * 4);
  float*  xg     = (float*)alloc((size_t)4096 * 1024 * 4);
  float*  h_seq  = (float*)alloc((size_t)4096 * 256 * 4);
  float*  a_res  = (float*)alloc((size_t)4096 * 256 * 4);
  float*  f_tmp  = (float*)alloc((size_t)4096 * 256 * 4);
  __bf16* Wf2    = (__bf16*)alloc((size_t)257 * 16 * 512 * 2);
  __bf16* Wih    = (__bf16*)alloc((size_t)8 * 64 * 512 * 2);
  __bf16* Whh    = (__bf16*)alloc((size_t)8 * 64 * 512 * 2);
  __bf16* L1f    = (__bf16*)alloc((size_t)8 * 16 * 512 * 2);
  __bf16* L3f    = (__bf16*)alloc((size_t)8 * 16 * 512 * 2);
  (void)ws_size; (void)in_sizes; (void)n_in; (void)out_size;

  // --- one-time (per launch) weight swizzles to WMMA fragment layout ---
  swizzle_bf16<<<8192, 256, 0, stream>>>(f2w, Wf2, 8192, 256);
  swizzle_bf16<<<32, 256, 0, stream>>>(f2b, Wf2 + (size_t)256 * 16 * 512, 32, 256);
  swizzle_bf16<<<1024, 256, 0, stream>>>(wih, Wih, 256, 1024);
  swizzle_bf16<<<1024, 256, 0, stream>>>(whh, Whh, 256, 1024);
  swizzle_bf16<<<256, 256, 0, stream>>>(l1_w, L1f, 256, 256);
  swizzle_bf16<<<256, 256, 0, stream>>>(l3_w, L3f, 256, 256);

  // --- per-asset embedding matvecs ---
  prep_assets<<<32, 256, 0, stream>>>(sid, vsn_emb, ws_w, ws_b, emb, h0_w1,
                                      h0_b1, h0_w2, h0_b2, c0_w1, c0_b1, c0_w2,
                                      c0_b2, ffn_emb, l2_w, l2_b, ev_ws, h0w,
                                      c0w, efw);

  // --- VSN gating softmax ---
  gate_softmax<<<4096, 256, 0, stream>>>(x, ev_ws, wx_w, wx_b, wout_w, wout_b,
                                         w_sm);

  // --- dominant fused VSN GEMM (17.2 GFLOP) + LayerNorm ---
  vsn_gemm<<<128, 256, 0, stream>>>(x, w_sm, f1w, f1b, Wf2, vnorm_g, vnorm_b,
                                    x_p);

  // --- LSTM input gates: xg = x_p @ Wih + bih ---
  gemm_k256<0><<<dim3(128, 4), 256, 0, stream>>>(x_p, Wih, 1024, bih, nullptr,
                                                 nullptr, nullptr, nullptr, xg);

  // --- LSTM recurrence (serial over T, one persistent workgroup) ---
  lstm_scan<<<1, 512, 0, stream>>>(Whh, xg, bhh, h0w, c0w, h_seq);

  // --- a = LN(h_seq + x_p) ---
  residual_ln<<<4096, 256, 0, stream>>>(h_seq, x_p, n1_g, n1_b, a_res);

  // --- FFN: hidden = elu(a@l1 + b1 + ef), out = LN(hidden@l3 + b3 + a) ---
  gemm_k256<1><<<dim3(128, 1), 256, 0, stream>>>(a_res, L1f, 256, l1_b, efw,
                                                 nullptr, nullptr, nullptr,
                                                 f_tmp);
  gemm_k256<2><<<dim3(128, 1), 256, 0, stream>>>(f_tmp, L3f, 256, l3_b, nullptr,
                                                 a_res, n2_g, n2_b,
                                                 (float*)d_out);
}
